// RSTB_56367150793361
// MI455X (gfx1250) — matrix operand order repaired and checked by reading, not compiled, not run on hardware
//
#include <hip/hip_runtime.h>
#include <math.h>

typedef __attribute__((ext_vector_type(8)))  float  v8f;
typedef __attribute__((ext_vector_type(16))) __bf16 v16bf;
typedef __attribute__((ext_vector_type(8)))  __bf16 v8bf;
typedef __attribute__((ext_vector_type(4)))  __bf16 v4bf;
typedef __attribute__((ext_vector_type(4)))  int    v4i;
typedef __attribute__((address_space(1)))    v4i    gv4i;   // global int4
typedef __attribute__((address_space(3)))    v4i    lv4i;   // LDS int4

#define IMG 512
#define P 4
#define CIN 3
#define E 96
#define NH 4
#define HD 24
#define WS 8
#define HP 128
#define NWIN 2048            // 8 * 16 * 16
#define NTOK 131072          // 8 * 128 * 128
#define SCALE 0.2041241452319315f  // 24^-0.5

#if __has_builtin(__builtin_amdgcn_global_load_async_to_lds_b128)
#define HAS_ASYNC 1
#else
#define HAS_ASYNC 0
#endif

#if HAS_ASYNC
#define WAIT_ASYNC() asm volatile("s_wait_asynccnt 0" ::: "memory")
#else
#define WAIT_ASYNC()
#endif

static __device__ __forceinline__ __bf16 tobf(float f) { return (__bf16)f; }

// async (or fallback) copy of 16 bytes (8 bf16) global -> LDS
static __device__ __forceinline__ void async_cp16(const __bf16* g, __bf16* l) {
#if HAS_ASYNC
  __builtin_amdgcn_global_load_async_to_lds_b128(
      (gv4i*)(void*)g, (lv4i*)(void*)l, 0, 0);
#else
  *(v8bf*)l = *(const v8bf*)g;
#endif
}

static __device__ __forceinline__ v16bf cat16(v8bf lo, v8bf hi) {
  return __builtin_shufflevector(lo, hi, 0, 1, 2, 3, 4, 5, 6, 7,
                                 8, 9, 10, 11, 12, 13, 14, 15);
}

// region label along one axis for the shifted-window attention mask (ws=8, ss=4)
static __device__ __forceinline__ int lab(int r) {
  return (r < HP - WS) ? 0 : ((r < HP - WS / 2) ? 1 : 2);
}

// ---------------------------------------------------------------------------
// Patch embed: t[b,hy,hx,e] = sum_{c,p,q} x[b,c,4hy+p,4hx+q] * w[e,c,p,q] + b[e]
// ---------------------------------------------------------------------------
__global__ __launch_bounds__(256) void patch_embed_k(
    const float* __restrict__ x, const float* __restrict__ w,
    const float* __restrict__ b, float* __restrict__ t) {
  int idx = blockIdx.x * 256 + threadIdx.x;           // NTOK * E
  int e = idx % E, tok = idx / E;
  int bb = tok >> 14, rem = tok & 16383;
  int hy = rem >> 7, hx = rem & 127;
  float acc = b[e];
  #pragma unroll
  for (int c = 0; c < CIN; ++c)
    #pragma unroll
    for (int p = 0; p < P; ++p)
      #pragma unroll
      for (int q = 0; q < P; ++q)
        acc += x[(((size_t)(bb * CIN + c) * IMG) + hy * P + p) * IMG + hx * P + q]
             * w[((e * CIN + c) * P + p) * P + q];
  t[(size_t)idx] = acc;
}

// ---------------------------------------------------------------------------
// LayerNorm -> bf16; windowed==1: gather into window order with cyclic shift
// ---------------------------------------------------------------------------
__global__ __launch_bounds__(256) void ln_k(
    const float* __restrict__ in, const float* __restrict__ g,
    const float* __restrict__ bta, __bf16* __restrict__ out,
    int shift, int windowed) {
  int wave = threadIdx.x >> 5, lane = threadIdx.x & 31;
  int row = blockIdx.x * 8 + wave;
  int b, y, x;
  if (windowed) {
    int win = row >> 6, tok = row & 63;
    b = win >> 8; int wrem = win & 255;
    int wy = wrem >> 4, wx = wrem & 15;
    y = (wy * WS + (tok >> 3) + shift) & 127;
    x = (wx * WS + (tok & 7) + shift) & 127;
  } else {
    b = row >> 14; int rem = row & 16383;
    y = rem >> 7; x = rem & 127;
  }
  const float* src = in + ((size_t)(b * HP + y) * HP + x) * E;
  float v0 = src[lane], v1 = src[lane + 32], v2 = src[lane + 64];
  float s = v0 + v1 + v2;
  for (int m = 1; m <= 16; m <<= 1) s += __shfl_xor(s, m, 32);
  float mean = s * (1.0f / 96.0f);
  float d0 = v0 - mean, d1 = v1 - mean, d2 = v2 - mean;
  float q = d0 * d0 + d1 * d1 + d2 * d2;
  for (int m = 1; m <= 16; m <<= 1) q += __shfl_xor(q, m, 32);
  float rstd = rsqrtf(q * (1.0f / 96.0f) + 1e-5f);
  __bf16* dst = out + (size_t)row * E;
  dst[lane]      = tobf(d0 * rstd * g[lane]      + bta[lane]);
  dst[lane + 32] = tobf(d1 * rstd * g[lane + 32] + bta[lane + 32]);
  dst[lane + 64] = tobf(d2 * rstd * g[lane + 64] + bta[lane + 64]);
}

// ---------------------------------------------------------------------------
// WMMA GEMM: C[R,N] = A[R,K] * W[K,N] + bias ; fuse: 0=none 1=gelu 2=+res
// Block tile 128x32; 8 waves, each wave 32x16 (2 WMMAs/K-step).
// Double-buffered LDS; AT=__bf16 A-tiles staged via async LDS DMA overlapped
// with the WMMAs of the previous K-step. W staged transposed so both fragment
// types load as contiguous 16B ds reads.
// ---------------------------------------------------------------------------
template <typename AT, typename OT>
__global__ __launch_bounds__(256) void gemm_k(
    const AT* __restrict__ A, const float* __restrict__ W,
    const float* __restrict__ bias, OT* __restrict__ C,
    const float* __restrict__ res, int K, int N, int fuse) {
  __shared__ alignas(16) __bf16 As[2][128][40];   // rows x k (stride 80B)
  __shared__ alignas(16) __bf16 BsT[2][32][40];   // cols x k (stride 80B)
  int tid = threadIdx.x;
  int lane = tid & 31, wave = tid >> 5;
  int half = lane >> 4, l15 = lane & 15;
  int wr = wave >> 1, wc = wave & 1;              // wave rows wr*32, cols wc*16
  int row0 = blockIdx.y * 128, col0 = blockIdx.x * 32;
  int arow = tid >> 1, ahh = (tid & 1) * 16;      // bf16 A staging assignment
  int brow = tid >> 3, bc4 = tid & 7;             // B staging assignment

  auto stage = [&](int buf, int k0) {
    if constexpr (sizeof(AT) == 2) {
      const __bf16* gp = (const __bf16*)A + (size_t)(row0 + arow) * K + k0 + ahh;
      async_cp16(gp,     &As[buf][arow][ahh]);
      async_cp16(gp + 8, &As[buf][arow][ahh + 8]);
    } else {
      #pragma unroll
      for (int j = 0; j < 4; ++j) {
        int f4 = tid + j * 256;                   // 0..1023
        int r = f4 >> 3, c4 = f4 & 7;
        float4 v = ((const float4*)((const float*)A + (size_t)(row0 + r) * K + k0))[c4];
        v4bf p; p[0] = tobf(v.x); p[1] = tobf(v.y); p[2] = tobf(v.z); p[3] = tobf(v.w);
        *(v4bf*)&As[buf][r][c4 * 4] = p;
      }
    }
    float4 v = ((const float4*)(W + (size_t)(k0 + brow) * N + col0))[bc4];
    BsT[buf][bc4 * 4 + 0][brow] = tobf(v.x);
    BsT[buf][bc4 * 4 + 1][brow] = tobf(v.y);
    BsT[buf][bc4 * 4 + 2][brow] = tobf(v.z);
    BsT[buf][bc4 * 4 + 3][brow] = tobf(v.w);
  };

  v8f acc0 = {}, acc1 = {};
  stage(0, 0);
  WAIT_ASYNC();
  __syncthreads();
  int cur = 0;
  for (int k0 = 0; k0 < K; k0 += 32) {
    if (k0 + 32 < K) stage(cur ^ 1, k0 + 32);     // DMA next tile during compute
    v16bf b = cat16(*(const v8bf*)&BsT[cur][wc * 16 + l15][half * 8],
                    *(const v8bf*)&BsT[cur][wc * 16 + l15][16 + half * 8]);
    v16bf a0 = cat16(*(const v8bf*)&As[cur][wr * 32 + l15][half * 8],
                     *(const v8bf*)&As[cur][wr * 32 + l15][16 + half * 8]);
    v16bf a1 = cat16(*(const v8bf*)&As[cur][wr * 32 + 16 + l15][half * 8],
                     *(const v8bf*)&As[cur][wr * 32 + 16 + l15][16 + half * 8]);
    acc0 = __builtin_amdgcn_wmma_f32_16x16x32_bf16(false, a0, false, b,
                                                   (short)0, acc0, false, false);
    acc1 = __builtin_amdgcn_wmma_f32_16x16x32_bf16(false, a1, false, b,
                                                   (short)0, acc1, false, false);
    WAIT_ASYNC();
    __syncthreads();
    cur ^= 1;
  }
  int n = col0 + wc * 16 + l15;
  float bn = bias[n];
  #pragma unroll
  for (int sub = 0; sub < 2; ++sub) {
    v8f acc = sub ? acc1 : acc0;
    #pragma unroll
    for (int r = 0; r < 8; ++r) {
      int m = row0 + wr * 32 + sub * 16 + half * 8 + r;
      float v = acc[r] + bn;
      if (fuse == 1)      v = 0.5f * v * (1.0f + erff(v * 0.7071067811865476f));
      else if (fuse == 2) v += res[(size_t)m * N + n];
      if constexpr (sizeof(OT) == 2) C[(size_t)m * N + n] = tobf(v);
      else                           C[(size_t)m * N + n] = v;
    }
  }
}

// ---------------------------------------------------------------------------
// Fused window attention (bf16 qkv in, bf16 out).
// Q/K staged via async DMA, padded to 32 dims; V transposed for B-frags.
// ---------------------------------------------------------------------------
__global__ __launch_bounds__(256) void win_attn_k(
    const __bf16* __restrict__ qkv, const float* __restrict__ rpb,
    __bf16* __restrict__ O, int shift) {
  __shared__ alignas(16) __bf16 qs[NH][64][32];   // 16 KB, dims 24..31 = 0
  __shared__ alignas(16) __bf16 ks_[NH][64][32];  // 16 KB
  __shared__ alignas(16) __bf16 vsT[NH][HD][64];  // 12 KB, [h][d][token]
  __shared__ alignas(16) __bf16 ps[8][16][64];    // 16 KB probs per wave

  int win = blockIdx.x, tid = threadIdx.x;
  const __bf16* base = qkv + (size_t)win * 64 * (3 * E);
  {
    // one (head, token) row per thread: q/k/v rows are 24 bf16 = 3x16B
    int h = tid >> 6, tk = tid & 63;
    const __bf16* qrow = base + (size_t)tk * (3 * E) + h * HD;
    const __bf16* krow = qrow + E;
    const __bf16* vrow = qrow + 2 * E;
    async_cp16(qrow,      &qs[h][tk][0]);
    async_cp16(qrow + 8,  &qs[h][tk][8]);
    async_cp16(qrow + 16, &qs[h][tk][16]);
    async_cp16(krow,      &ks_[h][tk][0]);
    async_cp16(krow + 8,  &ks_[h][tk][8]);
    async_cp16(krow + 16, &ks_[h][tk][16]);
    v8bf z = {};
    *(v8bf*)&qs[h][tk][24]  = z;   // pad dims 24..31
    *(v8bf*)&ks_[h][tk][24] = z;
    // V: 3 x b128 global loads, transposed 16-bit LDS stores
    v8bf v0 = *(const v8bf*)(vrow);
    v8bf v1 = *(const v8bf*)(vrow + 8);
    v8bf v2 = *(const v8bf*)(vrow + 16);
    #pragma unroll
    for (int d = 0; d < 8; ++d) {
      vsT[h][d][tk]      = v0[d];
      vsT[h][d + 8][tk]  = v1[d];
      vsT[h][d + 16][tk] = v2[d];
    }
  }
  WAIT_ASYNC();
  __syncthreads();

  int lane = tid & 31, wave = tid >> 5;
  int half = lane >> 4, l15 = lane & 15;
  int wrem = win & 255, wy = wrem >> 4, wx = wrem & 15;

  for (int pass = 0; pass < 2; ++pass) {
    int task = wave + pass * 8;
    int h = task >> 2, rt = task & 3;

    v16bf aq = cat16(*(const v8bf*)&qs[h][rt * 16 + l15][half * 8],
                     *(const v8bf*)&qs[h][rt * 16 + l15][16 + half * 8]);
    float val[4][8];
    #pragma unroll
    for (int t = 0; t < 4; ++t) {
      v16bf bk = cat16(*(const v8bf*)&ks_[h][t * 16 + l15][half * 8],
                       *(const v8bf*)&ks_[h][t * 16 + l15][16 + half * 8]);
      v8f c = {};
      c = __builtin_amdgcn_wmma_f32_16x16x32_bf16(false, aq, false, bk,
                                                  (short)0, c, false, false);
      int n = t * 16 + l15;
      #pragma unroll
      for (int r = 0; r < 8; ++r) {
        int m = rt * 16 + half * 8 + r;
        int dy = (m >> 3) - (n >> 3), dx = (m & 7) - (n & 7);
        float v = c[r] * SCALE + rpb[((dy + 7) * 15 + (dx + 7)) * NH + h];
        if (shift) {
          int la = lab(wy * WS + (m >> 3)) * 3 + lab(wx * WS + (m & 7));
          int lb = lab(wy * WS + (n >> 3)) * 3 + lab(wx * WS + (n & 7));
          if (la != lb) v -= 100.0f;
        }
        val[t][r] = v;
      }
    }
    // rowwise softmax over 64 cols (4 tiles x 16 lanes of this half)
    #pragma unroll
    for (int r = 0; r < 8; ++r) {
      float mx = fmaxf(fmaxf(val[0][r], val[1][r]), fmaxf(val[2][r], val[3][r]));
      for (int m2 = 1; m2 <= 8; m2 <<= 1) mx = fmaxf(mx, __shfl_xor(mx, m2, 32));
      float s = 0.0f;
      #pragma unroll
      for (int t = 0; t < 4; ++t) { val[t][r] = __expf(val[t][r] - mx); s += val[t][r]; }
      for (int m2 = 1; m2 <= 8; m2 <<= 1) s += __shfl_xor(s, m2, 32);
      float inv = 1.0f / s;
      #pragma unroll
      for (int t = 0; t < 4; ++t) val[t][r] *= inv;
    }
    #pragma unroll
    for (int t = 0; t < 4; ++t)
      #pragma unroll
      for (int r = 0; r < 8; ++r)
        ps[wave][half * 8 + r][t * 16 + l15] = tobf(val[t][r]);
    __syncthreads();

    // O(16x24) = P(16x64) x V(64x24): two col tiles, two K steps
    #pragma unroll
    for (int nt = 0; nt < 2; ++nt) {
      int d = nt * 16 + l15;
      v8f oc = {};
      #pragma unroll
      for (int ks2 = 0; ks2 < 2; ++ks2) {
        v16bf ap = cat16(*(const v8bf*)&ps[wave][l15][ks2 * 32 + half * 8],
                         *(const v8bf*)&ps[wave][l15][ks2 * 32 + 16 + half * 8]);
        v16bf bv;
        if (d < HD) {
          bv = cat16(*(const v8bf*)&vsT[h][d][ks2 * 32 + half * 8],
                     *(const v8bf*)&vsT[h][d][ks2 * 32 + 16 + half * 8]);
        } else {
          v16bf z = {}; bv = z;
        }
        oc = __builtin_amdgcn_wmma_f32_16x16x32_bf16(false, ap, false, bv,
                                                     (short)0, oc, false, false);
      }
      if (d < HD) {
        #pragma unroll
        for (int r = 0; r < 8; ++r) {
          int m = rt * 16 + half * 8 + r;
          O[((size_t)win * 64 + m) * E + h * HD + d] = tobf(oc[r]);
        }
      }
    }
    __syncthreads();
  }
}

// ---------------------------------------------------------------------------
// Scatter window-ordered proj output back to grid with inverse shift, += t
// ---------------------------------------------------------------------------
__global__ __launch_bounds__(256) void scatter_add_k(
    const float* __restrict__ pbuf, float* __restrict__ t, int shift) {
  int idx = blockIdx.x * 256 + threadIdx.x;   // NTOK * E
  int e = idx % E, row = idx / E;
  int win = row >> 6, tok = row & 63;
  int b = win >> 8; int wrem = win & 255;
  int wy = wrem >> 4, wx = wrem & 15;
  int y = (wy * WS + (tok >> 3) + shift) & 127;
  int x = (wx * WS + (tok & 7) + shift) & 127;
  t[((size_t)(b * HP + y) * HP + x) * E + e] += pbuf[(size_t)idx];
}

// ---------------------------------------------------------------------------
// Deconv (token -> 4x4 patch) + deconv bias + input residual
// ---------------------------------------------------------------------------
__global__ __launch_bounds__(256) void deconv_k(
    const float* __restrict__ f, const float* __restrict__ w,
    const float* __restrict__ b, const float* __restrict__ x,
    float* __restrict__ out) {
  int idx = blockIdx.x * 256 + threadIdx.x;   // 8*3*512*512
  int ix = idx & 511; int t1 = idx >> 9;
  int iy = t1 & 511;  int t2 = t1 >> 9;
  int c = t2 % CIN;   int bb = t2 / CIN;
  int hy = iy >> 2, p = iy & 3, hx = ix >> 2, q = ix & 3;
  const float* tok = f + ((size_t)(bb * HP + hy) * HP + hx) * E;
  float acc = b[c];
  #pragma unroll
  for (int e = 0; e < E; ++e)
    acc += tok[e] * w[((e * CIN + c) * P + p) * P + q];
  out[(size_t)idx] = acc + x[(size_t)idx];
}

// ---------------------------------------------------------------------------
extern "C" void kernel_launch(void* const* d_in, const int* in_sizes, int n_in,
                              void* d_out, int out_size, void* d_ws, size_t ws_size,
                              hipStream_t stream) {
  const float* x       = (const float*)d_in[0];
  const float* conv_w  = (const float*)d_in[1];
  const float* conv_b  = (const float*)d_in[2];
  const float* ln1_g   = (const float*)d_in[3];
  const float* ln1_b   = (const float*)d_in[4];
  const float* qkv_w   = (const float*)d_in[5];
  const float* qkv_b   = (const float*)d_in[6];
  const float* rpb     = (const float*)d_in[7];
  const float* proj_w  = (const float*)d_in[8];
  const float* proj_b  = (const float*)d_in[9];
  const float* ln2_g   = (const float*)d_in[10];
  const float* ln2_b   = (const float*)d_in[11];
  const float* fc1_w   = (const float*)d_in[12];
  const float* fc1_b   = (const float*)d_in[13];
  const float* fc2_w   = (const float*)d_in[14];
  const float* fc2_b   = (const float*)d_in[15];
  const float* lp_w    = (const float*)d_in[16];
  const float* lp_b    = (const float*)d_in[17];
  const float* dcv_w   = (const float*)d_in[18];
  const float* dcv_b   = (const float*)d_in[19];

  float* ws = (float*)d_ws;
  const size_t TSZ = (size_t)NTOK * E;      // 12.58M floats
  float*  tbuf = ws;                        // grid activations [B,HP,HP,E] f32
  __bf16* wbuf = (__bf16*)(ws + TSZ);       // LN1 windowed     [NWIN,64,E]
  __bf16* qkvb = (__bf16*)(ws + 2 * TSZ);   // qkv              [NWIN,64,3E]
  __bf16* obuf = (__bf16*)(ws + 5 * TSZ);   // attn out         [NWIN,64,E]
  float*  pbuf = ws + 6 * TSZ;              // proj out f32     [NWIN,64,E]
  __bf16* lbuf = (__bf16*)(ws + 7 * TSZ);   // LN2              [NTOK,E]
  __bf16* hbuf = (__bf16*)(ws + 8 * TSZ);   // MLP hidden       [NTOK,4E]
  float*  fbuf = ws + 12 * TSZ;             // final proj f32   [NTOK,E]

  patch_embed_k<<<(NTOK * E) / 256, 256, 0, stream>>>(x, conv_w, conv_b, tbuf);

  for (int i = 0; i < 4; ++i) {
    int shift = (i & 1) ? (WS / 2) : 0;
    ln_k<<<NTOK / 512, 256, 0, stream>>>(tbuf, ln1_g + i * E, ln1_b + i * E,
                                         wbuf, shift, 1);
    gemm_k<__bf16, __bf16><<<dim3(9, NTOK / 128), 256, 0, stream>>>(
        wbuf, qkv_w + (size_t)i * E * 3 * E, qkv_b + i * 3 * E, qkvb, nullptr,
        E, 3 * E, 0);
    win_attn_k<<<NWIN, 256, 0, stream>>>(qkvb, rpb + (size_t)i * 225 * NH,
                                         obuf, shift);
    gemm_k<__bf16, float><<<dim3(3, NTOK / 128), 256, 0, stream>>>(
        obuf, proj_w + (size_t)i * E * E, proj_b + i * E, pbuf, nullptr,
        E, E, 0);
    scatter_add_k<<<(NTOK * E) / 256, 256, 0, stream>>>(pbuf, tbuf, shift);
    ln_k<<<NTOK / 512, 256, 0, stream>>>(tbuf, ln2_g + i * E, ln2_b + i * E,
                                         lbuf, 0, 0);
    gemm_k<__bf16, __bf16><<<dim3(12, NTOK / 128), 256, 0, stream>>>(
        lbuf, fc1_w + (size_t)i * E * 4 * E, fc1_b + i * 4 * E, hbuf, nullptr,
        E, 4 * E, 1);
    gemm_k<__bf16, float><<<dim3(3, NTOK / 128), 256, 0, stream>>>(
        hbuf, fc2_w + (size_t)i * 4 * E * E, fc2_b + i * E, tbuf, tbuf,
        4 * E, E, 2);
  }

  gemm_k<float, float><<<dim3(3, NTOK / 128), 256, 0, stream>>>(
      tbuf, lp_w, lp_b, fbuf, nullptr, E, E, 0);
  deconv_k<<<(8 * CIN * IMG * IMG) / 256, 256, 0, stream>>>(
      fbuf, dcv_w, dcv_b, x, (float*)d_out);
}
